// NATTEN1DAttention_67929202754075
// MI455X (gfx1250) — matrix operand order repaired and checked by reading, not compiled, not run on hardware
//
#include <hip/hip_runtime.h>

typedef __attribute__((ext_vector_type(16))) _Float16 v16h;
typedef __attribute__((ext_vector_type(8)))  _Float16 v8h;
typedef __attribute__((ext_vector_type(4)))  _Float16 v4h;
typedef __attribute__((ext_vector_type(8)))  float    v8f;

namespace {
constexpr int Bn  = 2;
constexpr int Ln  = 2048;
constexpr int Dn  = 512;
constexpr int Hn  = 8;
constexpr int HD  = 64;    // head dim
constexpr int KS  = 63;    // kernel size
constexpr int KW  = 80;    // staged key-window rows (>= 15 + 63, multiple of 16)
constexpr int QS  = 72;    // Q LDS row stride (halves), padded vs 64 for bank spread
constexpr int KVS = 72;    // KV LDS row stride (halves)
constexpr int KTS = 104;   // transposed-KV LDS row stride (halves), >= 96
constexpr int PS  = 104;   // P LDS row stride (halves), >= 96
constexpr int WAVES = 2;   // heads (waves) per block -> 59.5 KB static LDS
}

__global__ __launch_bounds__(WAVES * 32) void
natten1d_wmma_kernel(const float* __restrict__ x, float* __restrict__ out) {
  __shared__ alignas(16) _Float16 sQ [WAVES][16 * QS];
  __shared__ alignas(16) _Float16 sKV[WAVES][KW * KVS];
  __shared__ alignas(16) _Float16 sKT[WAVES][HD * KTS];
  __shared__ alignas(16) _Float16 sP [WAVES][16 * PS];

  const int lane = threadIdx.x & 31;
  const int wv   = threadIdx.x >> 5;

  int bid = blockIdx.x;
  const int hp = bid % (Hn / WAVES);
  bid /= (Hn / WAVES);
  const int lt = bid % (Ln / 16);
  const int b  = bid / (Ln / 16);

  const int h     = hp * WAVES + wv;
  const int l0    = lt * 16;
  const int kbase = min(max(l0 - 31, 0), Ln - KS);  // union-window start for this tile

  const float* __restrict__ xb = x + (size_t)b * Ln * Dn + h * HD;

  _Float16* Q  = sQ[wv];
  _Float16* KV = sKV[wv];
  _Float16* KT = sKT[wv];
  _Float16* P  = sP[wv];

  // ---------------- Phase 1: stage Q (16xHD) and KV window (KWxHD) as f16 ----
  #pragma unroll
  for (int it = 0; it < 8; ++it) {                 // 16 rows * 16 float4
    int i  = it * 32 + lane;
    int r  = i >> 4, c4 = i & 15;
    const float4 v = ((const float4*)(xb + (size_t)(l0 + r) * Dn))[c4];
    v4h p = {};
    p[0] = (_Float16)v.x; p[1] = (_Float16)v.y;
    p[2] = (_Float16)v.z; p[3] = (_Float16)v.w;
    *(v4h*)&Q[r * QS + c4 * 4] = p;
  }
  #pragma unroll 4
  for (int it = 0; it < 40; ++it) {                // 80 rows * 16 float4
    int i   = it * 32 + lane;
    int r   = i >> 4, c4 = i & 15;
    int key = kbase + r;
    v4h p = {};
    if (key < Ln) {
      const float4 v = ((const float4*)(xb + (size_t)key * Dn))[c4];
      p[0] = (_Float16)v.x; p[1] = (_Float16)v.y;
      p[2] = (_Float16)v.z; p[3] = (_Float16)v.w;
    }
    *(v4h*)&KV[r * KVS + c4 * 4] = p;
  }
  __syncthreads();

  // ---------------- Phase 2: build transposed KV (channel x key) + zero pads --
  #pragma unroll 4
  for (int it = 0; it < 40; ++it) {
    int i  = it * 32 + lane;
    int r  = i >> 4, c0 = (i & 15) * 4;
    v4h v = *(const v4h*)&KV[r * KVS + c0];
    KT[(c0 + 0) * KTS + r] = v[0];
    KT[(c0 + 1) * KTS + r] = v[1];
    KT[(c0 + 2) * KTS + r] = v[2];
    KT[(c0 + 3) * KTS + r] = v[3];
  }
  {
    v4h z = {};
    #pragma unroll
    for (int it = 0; it < 8; ++it) {               // zero KT cols [80,96): 64 rows
      int i  = it * 32 + lane;
      int ch = i >> 2, c0 = (i & 3) * 4;
      *(v4h*)&KT[ch * KTS + KW + c0] = z;
    }
    #pragma unroll
    for (int it = 0; it < 2; ++it) {               // zero P cols [80,96): 16 rows
      int i = it * 32 + lane;
      int r = i >> 2, c0 = (i & 3) * 4;
      *(v4h*)&P[r * PS + KW + c0] = z;
    }
  }
  __syncthreads();

  const int m  = lane & 15;
  const int hs = lane >> 4;

  // ---------------- Phase 3: scores S = Q * K^T  (16 x 80), then softmax -----
  v16h qa[2];
  #pragma unroll
  for (int kc = 0; kc < 2; ++kc) {                 // A layout: lane=M, K 8-interleaved
    v8h lo = *(const v8h*)&Q[m * QS + kc * 32 + hs * 8];
    v8h hi = *(const v8h*)&Q[m * QS + kc * 32 + 16 + hs * 8];
    #pragma unroll
    for (int e = 0; e < 8; ++e) { qa[kc][e] = lo[e]; qa[kc][e + 8] = hi[e]; }
  }

  v8f acc[5];
  #pragma unroll
  for (int t = 0; t < 5; ++t) {
    v8f c = {};
    #pragma unroll
    for (int kc = 0; kc < 2; ++kc) {               // B layout: lane=N, K 8-interleaved
      const int row = t * 16 + m;                  // key row index in window
      v8h b0 = *(const v8h*)&KV[row * KVS + kc * 32 + hs * 8];
      v8h b1 = *(const v8h*)&KV[row * KVS + kc * 32 + 16 + hs * 8];
      v16h bb;
      #pragma unroll
      for (int e = 0; e < 8; ++e) { bb[e] = b0[e]; bb[e + 8] = b1[e]; }
      c = __builtin_amdgcn_wmma_f32_16x16x32_f16(false, qa[kc], false, bb,
                                                 (short)0, c, false, false);
    }
    acc[t] = c;
  }

  // softmax over window dim; C layout: row M = r + 8*hs, col N = m + 16*t
  #pragma unroll
  for (int r = 0; r < 8; ++r) {
    const int M   = r + 8 * hs;
    const int st  = min(max(l0 + M - 31, 0), Ln - KS);
    const int wlo = st - kbase;                    // this row's window start
    float sc[5], mx = -3.0e38f;
    #pragma unroll
    for (int t = 0; t < 5; ++t) {
      int  kwin  = m + 16 * t;
      bool valid = (kwin >= wlo) && (kwin < wlo + KS);
      sc[t] = valid ? acc[t][r] * 0.125f : -3.0e38f;   // scale = 64^-0.5
      mx = fmaxf(mx, sc[t]);
    }
    #pragma unroll
    for (int off = 8; off >= 1; off >>= 1) mx = fmaxf(mx, __shfl_xor(mx, off, 32));
    float ex[5], sum = 0.0f;
    #pragma unroll
    for (int t = 0; t < 5; ++t) {
      float e = (sc[t] > -1.0e38f) ? __expf(sc[t] - mx) : 0.0f;
      ex[t] = e; sum += e;
    }
    #pragma unroll
    for (int off = 8; off >= 1; off >>= 1) sum += __shfl_xor(sum, off, 32);
    const float inv = 1.0f / sum;
    #pragma unroll
    for (int t = 0; t < 5; ++t) P[M * PS + m + 16 * t] = (_Float16)(ex[t] * inv);
  }
  __syncthreads();

  // ---------------- Phase 4: out = P (16 x 96) * V (96 x 64) -----------------
  v16h pa[3];
  #pragma unroll
  for (int kc = 0; kc < 3; ++kc) {
    v8h lo = *(const v8h*)&P[m * PS + kc * 32 + hs * 8];
    v8h hi = *(const v8h*)&P[m * PS + kc * 32 + 16 + hs * 8];
    #pragma unroll
    for (int e = 0; e < 8; ++e) { pa[kc][e] = lo[e]; pa[kc][e + 8] = hi[e]; }
  }

  float* __restrict__ ob = out + (size_t)(b * Ln + l0) * Dn + h * HD;
  #pragma unroll
  for (int nt = 0; nt < 4; ++nt) {
    v8f c = {};
    #pragma unroll
    for (int kc = 0; kc < 3; ++kc) {
      const int ch = nt * 16 + m;                  // output channel for this lane
      v8h b0 = *(const v8h*)&KT[ch * KTS + kc * 32 + hs * 8];
      v8h b1 = *(const v8h*)&KT[ch * KTS + kc * 32 + 16 + hs * 8];
      v16h bb;
      #pragma unroll
      for (int e = 0; e < 8; ++e) { bb[e] = b0[e]; bb[e + 8] = b1[e]; }
      c = __builtin_amdgcn_wmma_f32_16x16x32_f16(false, pa[kc], false, bb,
                                                 (short)0, c, false, false);
    }
    #pragma unroll
    for (int r = 0; r < 8; ++r) {
      const int M = r + 8 * hs;
      ob[(size_t)M * Dn + nt * 16 + m] = c[r];     // 16 lanes -> 64B contiguous
    }
  }
}

extern "C" void kernel_launch(void* const* d_in, const int* in_sizes, int n_in,
                              void* d_out, int out_size, void* d_ws, size_t ws_size,
                              hipStream_t stream) {
  const float* x = (const float*)d_in[0];
  float* out = (float*)d_out;
  // num_heads (d_in[1]) and kernel_size (d_in[2]) are compile-time constants here.
  dim3 grid(Bn * (Ln / 16) * (Hn / WAVES));
  dim3 block(WAVES * 32);
  natten1d_wmma_kernel<<<grid, block, 0, stream>>>(x, out);
}
